// CosformerCausalAttention_64793876627783
// MI455X (gfx1250) — matrix operand order repaired and checked by reading, not compiled, not run on hardware
//
#include <hip/hip_runtime.h>
#include <hip/hip_bf16.h>

// ---------------------------------------------------------------------------
// Cosformer causal linear attention, chunked formulation, CDNA5 WMMA bf16
// + GLOBAL_LOAD_ASYNC_TO_LDS staging for the transpose-free operands.
// ---------------------------------------------------------------------------

typedef __bf16 bf16;
typedef __attribute__((ext_vector_type(8)))  bf16  v8bf;
typedef __attribute__((ext_vector_type(16))) bf16  v16bf;
typedef __attribute__((ext_vector_type(8)))  float v8f;

constexpr int Lq  = 2048;    // sequence length
constexpr int E   = 512;     // embed dim
constexpr int H   = 8;       // heads
constexpr int D   = 64;      // head dim
constexpr int E3  = 3 * E;   // 1536
constexpr int CH  = 64;      // chunk length
constexpr int NC  = Lq / CH; // 32 chunks

// --------------------------- fragment loaders ------------------------------
// A fragment: 16x32 bf16 tile, row-major source with given stride (elements).
// ISA layout: lane<16 -> row=lane, K = {0..7, 16..23}; lane>=16 -> row=lane-16,
// K = {8..15, 24..31}.  base points at tile origin (row0, k0).
__device__ inline v16bf load_a_frag(const bf16* base, int stride, int lane) {
    int r  = lane & 15;
    int kb = (lane >> 4) << 3;            // 0 or 8
    const bf16* p = base + r * stride + kb;
    v8bf lo = *(const v8bf*)(p);          // K kb .. kb+7
    v8bf hi = *(const v8bf*)(p + 16);     // K kb+16 .. kb+23
    return __builtin_shufflevector(lo, hi, 0,1,2,3,4,5,6,7,8,9,10,11,12,13,14,15);
}

// B fragment: 32x16 bf16 tile.  Source is B^T stored row-major [N][K] so each
// lane reads its column of B as two 16B runs.  bt points at (n0, k0) in B^T.
// ISA layout: lane<16 -> N=lane, K={0..7,16..23}; lane>=16 -> N=lane-16, K={8..15,24..31}.
__device__ inline v16bf load_b_frag(const bf16* bt, int stride, int lane) {
    int n  = lane & 15;
    int kb = (lane >> 4) << 3;            // 0 or 8
    const bf16* p = bt + n * stride + kb;
    v8bf lo = *(const v8bf*)(p);
    v8bf hi = *(const v8bf*)(p + 16);
    return __builtin_shufflevector(lo, hi, 0,1,2,3,4,5,6,7,8,9,10,11,12,13,14,15);
}

__device__ inline v8f wmma_bf16(v16bf a, v16bf b, v8f c) {
    // (neg_a, A, neg_b, B, c_mod, C, reuse_a, reuse_b)
    return __builtin_amdgcn_wmma_f32_16x16x32_bf16(false, a, false, b, (short)0, c,
                                                   false, false);
}

__device__ inline float poswt(int br, int l) {
    float ang = 1.5707963267948966f * (float)l / (float)Lq;
    return br == 0 ? __cosf(ang) : __sinf(ang);
}

// ---------------- gfx1250 async global->LDS copy (ASYNCcnt) ----------------
// Per-lane 16B copy: LDS[lds_off] = MEM[gsrc], tracked with ASYNCcnt.
__device__ inline void async_copy_b128(unsigned lds_off, const void* gsrc) {
    asm volatile("global_load_async_to_lds_b128 %0, %1, off"
                 :
                 : "v"(lds_off), "v"((unsigned long long)(uintptr_t)gsrc)
                 : "memory");
}
__device__ inline void async_wait0() {
    asm volatile("s_wait_asynccnt 0" ::: "memory");
}
// Generic LDS pointer low 32 bits == group-segment byte offset on AMDGPU.
__device__ inline unsigned lds_off32(const void* p) {
    return (unsigned)(uintptr_t)p;
}

// --------------------------- kernel 1: prep --------------------------------
// Convert x -> bf16; transpose-convert W_qkv, W_out to bf16 [N][K] layout.
__global__ void prep_kernel(const float* __restrict__ x,
                            const float* __restrict__ wqkv,
                            const float* __restrict__ wout,
                            bf16* __restrict__ xb,
                            bf16* __restrict__ wqkv_t,
                            bf16* __restrict__ wout_t) {
    int tid = blockIdx.x * blockDim.x + threadIdx.x;
    int nt  = gridDim.x * blockDim.x;
    for (int i = tid; i < Lq * E; i += nt) xb[i] = (bf16)x[i];
    for (int i = tid; i < E * E3; i += nt) {
        int k = i / E3, n = i % E3;
        wqkv_t[n * E + k] = (bf16)wqkv[i];
    }
    for (int i = tid; i < E * E; i += nt) {
        int k = i / E, n = i % E;
        wout_t[n * E + k] = (bf16)wout[i];
    }
}

// ------------------------ kernel 2: QKV projection -------------------------
// qkv = x @ W_qkv + b; relu on Q,K; per-head [h][l][d] fp32 buffers, plus a
// bf16 V^T [h*D][L] copy for async LDS staging downstream.
// 4 waves/block; each wave owns a 32(M) x 64(N) slab (8 accumulators).
__global__ __launch_bounds__(128) void qkv_gemm_kernel(
        const bf16* __restrict__ xb, const bf16* __restrict__ wt,
        const float* __restrict__ bias,
        float* __restrict__ Qp, float* __restrict__ Kp, float* __restrict__ Vv,
        bf16* __restrict__ VT) {
    int wave = threadIdx.x >> 5;
    int lane = threadIdx.x & 31;
    int m0 = blockIdx.y * 128 + wave * 32;
    int n0 = blockIdx.x * 64;
    v8f acc[2][4] = {};
    for (int k0 = 0; k0 < E; k0 += 32) {
        __builtin_prefetch(xb + (size_t)m0 * E + k0 + 32, 0, 3);
        v16bf a0 = load_a_frag(xb + (size_t)(m0)      * E + k0, E, lane);
        v16bf a1 = load_a_frag(xb + (size_t)(m0 + 16) * E + k0, E, lane);
#pragma unroll
        for (int t = 0; t < 4; ++t) {
            v16bf b = load_b_frag(wt + (size_t)(n0 + t * 16) * E + k0, E, lane);
            acc[0][t] = wmma_bf16(a0, b, acc[0][t]);
            acc[1][t] = wmma_bf16(a1, b, acc[1][t]);
        }
    }
    int hi8 = (lane >> 4) * 8;
#pragma unroll
    for (int r = 0; r < 2; ++r) {
#pragma unroll
        for (int t = 0; t < 4; ++t) {
            int n = n0 + t * 16 + (lane & 15);
            int which = n / E;
            int e = n % E;
            int h = e / D, d = e % D;
            float bia = bias[n];
#pragma unroll
            for (int i = 0; i < 8; ++i) {
                int m = m0 + r * 16 + i + hi8;
                float v = acc[r][t][i] + bia;
                if (which == 0)      Qp[(h * Lq + m) * D + d] = v > 0.f ? v : 0.f;
                else if (which == 1) Kp[(h * Lq + m) * D + d] = v > 0.f ? v : 0.f;
                else {
                    Vv[(h * Lq + m) * D + d] = v;
                    VT[(size_t)(h * D + d) * Lq + m] = (bf16)v;
                }
            }
        }
    }
}

// ----------------- kernel 3: per-chunk KV sums (Kw^T @ V) ------------------
__global__ __launch_bounds__(128) void chunk_sums_kernel(
        const float* __restrict__ Kp, const bf16* __restrict__ VT,
        float* __restrict__ Sc, float* __restrict__ Ksc) {
    int chunk = blockIdx.x, h = blockIdx.y, br = blockIdx.z;
    int lbase = chunk * CH;
    __shared__ __align__(16) bf16 sKwT[64][64];  // [d][l]
    __shared__ __align__(16) bf16 sVT[64][64];   // [d][l]
    int t = threadIdx.x;
    // async-stage V^T chunk: 64 rows x 128B, 16B per lane-op.
#pragma unroll
    for (int it = 0; it < 4; ++it) {
        int p  = it * 128 + t;          // 0..511 16B pieces
        int d  = p >> 3;
        int po = (p & 7) * 8;           // element offset within row
        async_copy_b128(lds_off32(&sVT[d][po]),
                        VT + (size_t)(h * D + d) * Lq + lbase + po);
    }
    // VALU-stage Kw^T (needs per-position cos/sin weight).
    for (int idx = t; idx < 4096; idx += 128) {
        int l = idx >> 6, d = idx & 63;
        int gl = lbase + l;
        sKwT[d][l] = (bf16)(Kp[(h * Lq + gl) * D + d] * poswt(br, gl));
    }
    async_wait0();
    __syncthreads();
    int wave = t >> 5, lane = t & 31;
    int m0 = wave * 16;
    v8f acc[4] = {};
    for (int k0 = 0; k0 < CH; k0 += 32) {
        v16bf a = load_a_frag(&sKwT[m0][k0], 64, lane);   // A[m=d][k=l]
#pragma unroll
        for (int tt = 0; tt < 4; ++tt) {
            v16bf b = load_b_frag(&sVT[tt * 16][k0], 64, lane); // B[k=l][n=e]
            acc[tt] = wmma_bf16(a, b, acc[tt]);
        }
    }
    float* S = Sc + (((size_t)(br * H + h) * NC + chunk) << 12);  // 64*64
    int hi8 = (lane >> 4) * 8;
#pragma unroll
    for (int tt = 0; tt < 4; ++tt) {
        int e = tt * 16 + (lane & 15);
#pragma unroll
        for (int i = 0; i < 8; ++i) {
            int d = m0 + i + hi8;
            S[d * 64 + e] = acc[tt][i];
        }
    }
    if (t < 64) {  // per-chunk k-sum (for the normalizer)
        float s = 0.f;
        for (int l = 0; l < 64; ++l) s += (float)sKwT[t][l];
        Ksc[((size_t)(br * H + h) * NC + chunk) * 64 + t] = s;
    }
}

// --------------- kernel 4: exclusive prefix over 32 chunks -----------------
// fp32 running sums; emits S_prefix^T directly in bf16 [e][d] layout so the
// consumer can async-copy it straight into its B^T LDS tile.
__global__ __launch_bounds__(256) void scan_kernel(
        const float* __restrict__ Sc, const float* __restrict__ Ksc,
        bf16* __restrict__ SpT, float* __restrict__ Ksp) {
    int bh = blockIdx.x;           // br*H + h  (16 blocks)
    int t = threadIdx.x;
    const int NEL = 4096 + 64;
    for (int el = t; el < NEL; el += 256) {
        float run = 0.f;
        for (int c = 0; c < NC; ++c) {
            if (el < 4096) {
                int d = el >> 6, e = el & 63;
                size_t base = (size_t)(bh * NC + c) << 12;
                SpT[base + (e << 6) + d] = (bf16)run;
                run += Sc[base + el];
            } else {
                int d = el - 4096;
                size_t o = (size_t)(bh * NC + c) * 64 + d;
                Ksp[o] = run;
                run += Ksc[o];
            }
        }
    }
}

// ------------- kernel 5: per-chunk attention (inter + intra) ---------------
__global__ __launch_bounds__(128) void chunk_attn_kernel(
        const float* __restrict__ Qp, const float* __restrict__ Kp,
        const bf16* __restrict__ VT, const bf16* __restrict__ SpT,
        const float* __restrict__ Ksp, bf16* __restrict__ attn) {
    int chunk = blockIdx.x, h = blockIdx.y;
    int lbase = chunk * CH;
    __shared__ __align__(16) bf16 sQ[64][64];    // Qw row-major [l][d]
    __shared__ __align__(16) bf16 sK[64][64];    // Kw row-major [l][d] (= B^T for P)
    __shared__ __align__(16) bf16 sVT[64][64];   // V^T [d][l]  (= B^T for P@V)
    __shared__ __align__(16) bf16 sST[64][64];   // S_prefix^T [e][d] (= B^T for Q@S)
    __shared__ __align__(16) bf16 sP[64][64];    // masked P [lq][lk]
    __shared__ float sNrm[64];
    __shared__ float sKsum[64];
    int t = threadIdx.x;
    int wave = t >> 5, lane = t & 31;
    int m0 = wave * 16;
    v8f ctx[4] = {};
    if (t < 64) sNrm[t] = 0.f;

    // V^T is branch-independent: async-stage it once.
#pragma unroll
    for (int it = 0; it < 4; ++it) {
        int p  = it * 128 + t;
        int d  = p >> 3;
        int po = (p & 7) * 8;
        async_copy_b128(lds_off32(&sVT[d][po]),
                        VT + (size_t)(h * D + d) * Lq + lbase + po);
    }

    for (int br = 0; br < 2; ++br) {
        __syncthreads();
        // async-stage S_prefix^T (already bf16, already transposed): 8KB.
        const bf16* SbT = SpT + (((size_t)(br * H + h) * NC + chunk) << 12);
#pragma unroll
        for (int it = 0; it < 4; ++it) {
            int elo = (it * 128 + t) * 8;   // 8 bf16 = 16B per lane-op
            async_copy_b128(lds_off32(&sST[0][0]) + elo * 2, SbT + elo);
        }
        // VALU-stage Qw, Kw (need cos/sin position weight).
        for (int idx = t; idx < 4096; idx += 128) {
            int l = idx >> 6, d = idx & 63;
            int gl = lbase + l;
            float w = poswt(br, gl);
            sQ[l][d] = (bf16)(Qp[(h * Lq + gl) * D + d] * w);
            sK[l][d] = (bf16)(Kp[(h * Lq + gl) * D + d] * w);
        }
        if (t < 64) sKsum[t] = Ksp[((size_t)(br * H + h) * NC + chunk) * 64 + t];
        async_wait0();
        __syncthreads();

        // ctx += Qw @ S_prefix   (inter-chunk)
        for (int k0 = 0; k0 < D; k0 += 32) {
            v16bf a = load_a_frag(&sQ[m0][k0], 64, lane);
#pragma unroll
            for (int tt = 0; tt < 4; ++tt) {
                v16bf b = load_b_frag(&sST[tt * 16][k0], 64, lane);
                ctx[tt] = wmma_bf16(a, b, ctx[tt]);
            }
        }
        // P = Qw @ Kw^T
        v8f pacc[4] = {};
        for (int k0 = 0; k0 < D; k0 += 32) {
            v16bf a = load_a_frag(&sQ[m0][k0], 64, lane);
#pragma unroll
            for (int tt = 0; tt < 4; ++tt) {
                v16bf b = load_b_frag(&sK[tt * 16][k0], 64, lane);
                pacc[tt] = wmma_bf16(a, b, pacc[tt]);
            }
        }
        // causal mask (keep key index n <= query index m) and park in LDS.
        int hi8 = (lane >> 4) * 8;
#pragma unroll
        for (int tt = 0; tt < 4; ++tt) {
            int n = tt * 16 + (lane & 15);
#pragma unroll
            for (int i = 0; i < 8; ++i) {
                int m = m0 + i + hi8;
                sP[m][n] = (bf16)(n <= m ? pacc[tt][i] : 0.f);
            }
        }
        __syncthreads();
        // normalizer: rowsum(Pmask) + Qw . ksum_prefix
        if (t < 64) {
            float s = 0.f;
            for (int n = 0; n < 64; ++n) s += (float)sP[t][n];
            for (int d = 0; d < 64; ++d) s += (float)sQ[t][d] * sKsum[d];
            sNrm[t] += s;
        }
        // ctx += Pmask @ V   (intra-chunk)
        for (int k0 = 0; k0 < CH; k0 += 32) {
            v16bf a = load_a_frag(&sP[m0][k0], 64, lane);
#pragma unroll
            for (int tt = 0; tt < 4; ++tt) {
                v16bf b = load_b_frag(&sVT[tt * 16][k0], 64, lane);
                ctx[tt] = wmma_bf16(a, b, ctx[tt]);
            }
        }
    }
    __syncthreads();
    int hi8 = (lane >> 4) * 8;
#pragma unroll
    for (int tt = 0; tt < 4; ++tt) {
        int d = tt * 16 + (lane & 15);
#pragma unroll
        for (int i = 0; i < 8; ++i) {
            int m = m0 + i + hi8;
            float v = ctx[tt][i] / (sNrm[m] + 1e-6f);
            attn[(size_t)(lbase + m) * E + h * D + d] = (bf16)v;
        }
    }
}

// ---------------------- kernel 6: output projection ------------------------
__global__ __launch_bounds__(128) void out_gemm_kernel(
        const bf16* __restrict__ attn, const bf16* __restrict__ wot,
        const float* __restrict__ bias, float* __restrict__ out) {
    int wave = threadIdx.x >> 5;
    int lane = threadIdx.x & 31;
    int m0 = blockIdx.y * 128 + wave * 32;
    int n0 = blockIdx.x * 64;
    v8f acc[2][4] = {};
    for (int k0 = 0; k0 < E; k0 += 32) {
        __builtin_prefetch(attn + (size_t)m0 * E + k0 + 32, 0, 3);
        v16bf a0 = load_a_frag(attn + (size_t)(m0)      * E + k0, E, lane);
        v16bf a1 = load_a_frag(attn + (size_t)(m0 + 16) * E + k0, E, lane);
#pragma unroll
        for (int tt = 0; tt < 4; ++tt) {
            v16bf b = load_b_frag(wot + (size_t)(n0 + tt * 16) * E + k0, E, lane);
            acc[0][tt] = wmma_bf16(a0, b, acc[0][tt]);
            acc[1][tt] = wmma_bf16(a1, b, acc[1][tt]);
        }
    }
    int hi8 = (lane >> 4) * 8;
#pragma unroll
    for (int r = 0; r < 2; ++r) {
#pragma unroll
        for (int tt = 0; tt < 4; ++tt) {
            int n = n0 + tt * 16 + (lane & 15);
            float bia = bias[n];
#pragma unroll
            for (int i = 0; i < 8; ++i) {
                int m = m0 + r * 16 + i + hi8;
                out[(size_t)m * E + n] = acc[r][tt][i] + bia;
            }
        }
    }
}

// ------------------------------ launcher -----------------------------------
extern "C" void kernel_launch(void* const* d_in, const int* in_sizes, int n_in,
                              void* d_out, int out_size, void* d_ws, size_t ws_size,
                              hipStream_t stream) {
    const float* x     = (const float*)d_in[0];
    const float* W_qkv = (const float*)d_in[1];
    const float* b_qkv = (const float*)d_in[2];
    const float* W_out = (const float*)d_in[3];
    const float* b_out = (const float*)d_in[4];
    float* out = (float*)d_out;

    char* ws = (char*)d_ws;
    size_t off = 0;
    auto alloc = [&](size_t bytes) { void* p = ws + off; off = (off + bytes + 255) & ~(size_t)255; return p; };
    bf16*  xb      = (bf16*) alloc((size_t)Lq * E * 2);
    bf16*  wqkv_t  = (bf16*) alloc((size_t)E3 * E * 2);
    bf16*  wout_t  = (bf16*) alloc((size_t)E  * E * 2);
    float* Qp      = (float*)alloc((size_t)H * Lq * D * 4);
    float* Kp      = (float*)alloc((size_t)H * Lq * D * 4);
    float* Vv      = (float*)alloc((size_t)H * Lq * D * 4);
    bf16*  VT      = (bf16*) alloc((size_t)H * D * Lq * 2);
    float* Sc      = (float*)alloc((size_t)2 * H * NC * 64 * 64 * 4);
    bf16*  SpT     = (bf16*) alloc((size_t)2 * H * NC * 64 * 64 * 2);
    float* Ksc     = (float*)alloc((size_t)2 * H * NC * 64 * 4);
    float* Ksp     = (float*)alloc((size_t)2 * H * NC * 64 * 4);
    bf16*  attn    = (bf16*) alloc((size_t)Lq * E * 2);
    (void)ws_size; (void)in_sizes; (void)n_in; (void)out_size; (void)Vv;

    prep_kernel<<<512, 256, 0, stream>>>(x, W_qkv, W_out, xb, wqkv_t, wout_t);

    qkv_gemm_kernel<<<dim3(E3 / 64, Lq / 128), 128, 0, stream>>>(
        xb, wqkv_t, b_qkv, Qp, Kp, Vv, VT);

    chunk_sums_kernel<<<dim3(NC, H, 2), 128, 0, stream>>>(Kp, VT, Sc, Ksc);

    scan_kernel<<<2 * H, 256, 0, stream>>>(Sc, Ksc, SpT, Ksp);

    chunk_attn_kernel<<<dim3(NC, H), 128, 0, stream>>>(Qp, Kp, VT, SpT, Ksp, attn);

    out_gemm_kernel<<<dim3(E / 64, Lq / 128), 128, 0, stream>>>(
        attn, wout_t, b_out, out);
}
